// GeneralCausalSelfAttention_74861279969486
// MI455X (gfx1250) — matrix operand order repaired and checked by reading, not compiled, not run on hardware
//
#include <hip/hip_runtime.h>
#include <stdint.h>

// ---------------------------------------------------------------------------
// Problem constants (reference: B=2, T=2048, C=1024, nh=16, hs=64)
// ---------------------------------------------------------------------------
#define BSZ  2
#define TSEQ 2048
#define CDIM 1024
#define NH   16
#define HS   64
#define MTOT (BSZ * TSEQ)     // 4096 rows
#define N2C  (2 * CDIM)       // 2048

typedef __attribute__((ext_vector_type(16))) __bf16 v16bf;
typedef __attribute__((ext_vector_type(8)))  float  v8f;

union FragAB { v16bf v; uint32_t u[8]; };

// round-to-nearest-even f32 -> bf16 (bit pattern)
static __device__ __forceinline__ uint16_t f2bf(float f) {
    union { float f; uint32_t u; } a; a.f = f;
    uint32_t r = a.u + 0x7FFFu + ((a.u >> 16) & 1u);
    return (uint16_t)(r >> 16);
}

static __device__ __forceinline__ v8f wmma_bf16(v16bf a, v16bf b, v8f c) {
    return __builtin_amdgcn_wmma_f32_16x16x32_bf16(
        /*neg_a=*/false, a, /*neg_b=*/false, b,
        /*c_mod=*/(short)0, c, /*reuse_a=*/false, /*reuse_b=*/false);
}

// Async copy: global -> LDS, 16 bytes per enabled lane. ASYNCcnt-tracked.
static __device__ __forceinline__ void async_cp_b128(uint32_t lds_off, uint64_t gaddr) {
    asm volatile("global_load_async_to_lds_b128 %0, %1, off"
                 :: "v"(lds_off), "v"(gaddr)
                 : "memory");
}
static __device__ __forceinline__ void wait_asynccnt0() {
    asm volatile("s_wait_asynccnt 0" ::: "memory");
}

// A fragment (16x32, row-major source, bf16). base points at (m0,k0).
// ISA layout: lanes0-15 M=lane, VGPRv K = (v<4 ? 2v : 2v+8); lanes16-31 +8 on K.
static __device__ __forceinline__ v16bf load_a_frag(const uint16_t* base, int lda, int lane) {
    FragAB f;
    int m  = lane & 15;
    int hf = lane >> 4;
#pragma unroll
    for (int v = 0; v < 8; ++v) {
        int k = (v < 4 ? 2 * v : 2 * v + 8) + hf * 8;
        f.u[v] = *(const uint32_t*)(base + m * lda + k);
    }
    return f.v;
}

// B fragment (32x16): source stored N-major (row n of length-K, stride ldb).
// ISA layout: lanes0-15 N=lane, VGPRv K = (v<4 ? 2v : 2v+8); lanes16-31 +8 on K.
static __device__ __forceinline__ v16bf load_b_frag(const uint16_t* base, int ldb, int lane) {
    FragAB f;
    int n  = lane & 15;
    int kb = (lane >> 4) * 8;
#pragma unroll
    for (int v = 0; v < 8; ++v) {
        f.u[v] = *(const uint32_t*)(base + n * ldb + kb + (v < 4 ? 2 * v : 2 * v + 8));
    }
    return f.v;
}

// ---------------------------------------------------------------------------
// Prep kernels
// ---------------------------------------------------------------------------
__global__ void k_convert_bf16(const float* __restrict__ x, uint16_t* __restrict__ o, int n) {
    int i = blockIdx.x * blockDim.x + threadIdx.x;
    if (i < n) o[i] = f2bf(x[i]);
}

// Wt[n*K + k] = bf16(W[k*N + n])   (K-major -> N-major)
__global__ void k_transpose_bf16(const float* __restrict__ W, uint16_t* __restrict__ Wt,
                                 int K, int N) {
    int i = blockIdx.x * blockDim.x + threadIdx.x;
    if (i < K * N) {
        int n = i / K;
        int k = i - n * K;
        Wt[i] = f2bf(W[k * N + n]);
    }
}

// ---------------------------------------------------------------------------
// Fused QKV GEMM: qk_tv = x @ W_ca + b_ca ; split into qk (row-major bf16)
// and v transposed per head: vt[b][h][d][t] (bf16, N-major for PV GEMM).
// The shared 16x32 A-tile is staged once per workgroup via
// global_load_async_to_lds_b128 (ASYNCcnt), then all 4 waves read fragments
// from LDS. grid: (MTOT/16, N2C/256), block: 128 (4 waves, 64 N-cols each)
// ---------------------------------------------------------------------------
__global__ __launch_bounds__(128) void k_gemm_qkv(const uint16_t* __restrict__ xbf,
                                                  const uint16_t* __restrict__ wt,   // [2048][1024] N-major
                                                  const float* __restrict__ bias,    // [2048]
                                                  uint16_t* __restrict__ qk,         // [B*T][C]
                                                  uint16_t* __restrict__ vt) {       // [B][NH][HS][T]
    __shared__ uint16_t Atile[16 * 32];   // 1 KB staging tile (rows m0..m0+15, k-step)

    int lane = threadIdx.x & 31;
    int wv   = threadIdx.x >> 5;
    int m0   = blockIdx.x * 16;
    int n0   = blockIdx.y * 256 + wv * 64;

    v8f acc[4] = {};
    for (int k0 = 0; k0 < CDIM; k0 += 32) {
        if (wv == 0) {
            // 16 rows x 64 B = 64 x b128 segments; 2 per lane
#pragma unroll
            for (int j = 0; j < 2; ++j) {
                int e   = lane + j * 32;
                int row = e >> 2;
                int seg = e & 3;                       // 16-byte segment
                uint64_t g = (uint64_t)(uintptr_t)(xbf + (m0 + row) * CDIM + k0 + seg * 8);
                uint32_t l = (uint32_t)(uintptr_t)&Atile[row * 32 + seg * 8];
                async_cp_b128(l, g);
            }
            wait_asynccnt0();
        }
        __syncthreads();

        v16bf a = load_a_frag(Atile, 32, lane);        // ds_load from LDS
#pragma unroll
        for (int nt = 0; nt < 4; ++nt) {
            v16bf b = load_b_frag(wt + (n0 + nt * 16) * CDIM + k0, CDIM, lane);
            acc[nt] = wmma_bf16(a, b, acc[nt]);
        }
        __syncthreads();
    }

    int laneN = lane & 15;
    int hf    = lane >> 4;
#pragma unroll
    for (int nt = 0; nt < 4; ++nt) {
        int   n  = n0 + nt * 16 + laneN;
        float bv = bias[n];
#pragma unroll
        for (int r = 0; r < 8; ++r) {
            int row = m0 + r + 8 * hf;            // global row in [0, B*T)
            float val = acc[nt][r] + bv;
            uint16_t h16 = f2bf(val);
            if (n < CDIM) {
                qk[row * CDIM + n] = h16;
            } else {
                int c = n - CDIM;
                int hh = c >> 6;
                int d  = c & 63;
                int bb = row >> 11;               // row / T
                int t  = row & (TSEQ - 1);
                vt[((bb * NH + hh) * HS + d) * TSEQ + t] = h16;
            }
        }
    }
}

// ---------------------------------------------------------------------------
// Flash attention: one wave per (b, h, 16-query tile), 64-key chunks.
// 16 WMMAs per online-softmax pass. grid: B*NH*(T/16) = 4096, block: 32
// ---------------------------------------------------------------------------
__global__ __launch_bounds__(32) void k_attn(const uint16_t* __restrict__ xbf,
                                             const uint16_t* __restrict__ qk,
                                             const uint16_t* __restrict__ vt,
                                             uint16_t* __restrict__ y) {
    __shared__ uint16_t Ptile[16 * 68];   // 16 q-rows x 64 keys, stride 68 (b32-aligned rows)

    int lane = threadIdx.x & 31;
    int bid  = blockIdx.x;
    int qt = bid & 127;
    int h  = (bid >> 7) & 15;
    int b  = bid >> 11;
    int q0 = qt * 16;

    // log2-domain softmax: p = exp2(s*scaleL2E - m2), m2 tracked in log2 units
    const float scaleL2E = 0.125f * 1.44269504f;   // (1/sqrt(64)) * log2(e)

    // x A-fragments for this query tile (16 x 64 = two K=32 fragments)
    const uint16_t* xbase = xbf + (b * TSEQ + q0) * CDIM + h * HS;
    v16bf a0 = load_a_frag(xbase,      CDIM, lane);
    v16bf a1 = load_a_frag(xbase + 32, CDIM, lane);

    v8f o[4] = {};
    float mrow[8], lrow[8];
#pragma unroll
    for (int r = 0; r < 8; ++r) { mrow[r] = -INFINITY; lrow[r] = 0.0f; }

    int laneN = lane & 15;
    int hf    = lane >> 4;
    int kmax  = q0 + 15;

    for (int k0 = 0; k0 <= kmax; k0 += 64) {
        // ---- scores: s = x_h @ qk_h^T  (four 16x16 key tiles, K=64) ----
        v8f s[4] = {};
        const uint16_t* kb0 = qk + (b * TSEQ + k0) * CDIM + h * HS;
#pragma unroll
        for (int nt = 0; nt < 4; ++nt) {
            v16bf bA = load_b_frag(kb0 + nt * 16 * CDIM,      CDIM, lane); // d 0..31
            v16bf bB = load_b_frag(kb0 + nt * 16 * CDIM + 32, CDIM, lane); // d 32..63
            s[nt] = wmma_bf16(a0, bA, s[nt]);
            s[nt] = wmma_bf16(a1, bB, s[nt]);
        }

        // ---- online softmax over this 64-key chunk (log2 domain) ----
#pragma unroll
        for (int r = 0; r < 8; ++r) {
            int q   = q0 + r + 8 * hf;
            int row = r + 8 * hf;
            float sv[4];
#pragma unroll
            for (int nt = 0; nt < 4; ++nt) {
                float v = s[nt][r] * scaleL2E;
                if (k0 + nt * 16 + laneN > q) v = -INFINITY;   // causal mask
                sv[nt] = v;
            }
            float cm = fmaxf(fmaxf(sv[0], sv[1]), fmaxf(sv[2], sv[3]));
            cm = fmaxf(cm, __shfl_xor(cm, 1, 16));
            cm = fmaxf(cm, __shfl_xor(cm, 2, 16));
            cm = fmaxf(cm, __shfl_xor(cm, 4, 16));
            cm = fmaxf(cm, __shfl_xor(cm, 8, 16));

            float mnew = fmaxf(mrow[r], cm);
            float corr = exp2f(mrow[r] - mnew);
            float p[4];
            float rs = 0.0f;
#pragma unroll
            for (int nt = 0; nt < 4; ++nt) {
                p[nt] = exp2f(sv[nt] - mnew);
                rs += p[nt];
            }
            rs += __shfl_xor(rs, 1, 16);
            rs += __shfl_xor(rs, 2, 16);
            rs += __shfl_xor(rs, 4, 16);
            rs += __shfl_xor(rs, 8, 16);

            lrow[r] = lrow[r] * corr + rs;
            mrow[r] = mnew;
#pragma unroll
            for (int dt = 0; dt < 4; ++dt) o[dt][r] *= corr;

#pragma unroll
            for (int nt = 0; nt < 4; ++nt)
                Ptile[row * 68 + nt * 16 + laneN] = f2bf(p[nt]);
        }
        __syncthreads();

        // ---- re-layout P (C-layout -> A-layout) and accumulate P @ V ----
        v16bf pa0 = load_a_frag(Ptile,      68, lane);   // keys k0..k0+31
        v16bf pa1 = load_a_frag(Ptile + 32, 68, lane);   // keys k0+32..k0+63
        const uint16_t* vb = vt + ((b * NH + h) * HS) * TSEQ + k0;
#pragma unroll
        for (int dt = 0; dt < 4; ++dt) {
            v16bf bv0 = load_b_frag(vb + dt * 16 * TSEQ,      TSEQ, lane);
            v16bf bv1 = load_b_frag(vb + dt * 16 * TSEQ + 32, TSEQ, lane);
            o[dt] = wmma_bf16(pa0, bv0, o[dt]);
            o[dt] = wmma_bf16(pa1, bv1, o[dt]);
        }
        __syncthreads();
    }

    // ---- normalize + store y (bf16 row-major [B][T][C]) ----
    float inv[8];
#pragma unroll
    for (int r = 0; r < 8; ++r) inv[r] = 1.0f / lrow[r];
#pragma unroll
    for (int dt = 0; dt < 4; ++dt) {
#pragma unroll
        for (int r = 0; r < 8; ++r) {
            int t   = q0 + r + 8 * hf;
            int col = h * HS + dt * 16 + laneN;
            y[(b * TSEQ + t) * CDIM + col] = f2bf(o[dt][r] * inv[r]);
        }
    }
}

// ---------------------------------------------------------------------------
// Output projection: out = y @ W_proj + b_proj (f32 out)
// grid: (MTOT/16, CDIM/256), block: 128
// ---------------------------------------------------------------------------
__global__ __launch_bounds__(128) void k_gemm_proj(const uint16_t* __restrict__ ybf,
                                                   const uint16_t* __restrict__ wtp,  // [1024][1024] N-major
                                                   const float* __restrict__ bias,
                                                   float* __restrict__ out) {
    int lane = threadIdx.x & 31;
    int wv   = threadIdx.x >> 5;
    int m0   = blockIdx.x * 16;
    int n0   = blockIdx.y * 256 + wv * 64;

    v8f acc[4] = {};
    for (int k0 = 0; k0 < CDIM; k0 += 32) {
        v16bf a = load_a_frag(ybf + m0 * CDIM + k0, CDIM, lane);
#pragma unroll
        for (int nt = 0; nt < 4; ++nt) {
            v16bf b = load_b_frag(wtp + (n0 + nt * 16) * CDIM + k0, CDIM, lane);
            acc[nt] = wmma_bf16(a, b, acc[nt]);
        }
    }

    int laneN = lane & 15;
    int hf    = lane >> 4;
#pragma unroll
    for (int nt = 0; nt < 4; ++nt) {
        int   n  = n0 + nt * 16 + laneN;
        float bv = bias[n];
#pragma unroll
        for (int r = 0; r < 8; ++r) {
            int row = m0 + r + 8 * hf;
            out[row * CDIM + n] = acc[nt][r] + bv;
        }
    }
}

// ---------------------------------------------------------------------------
// Launch
// ---------------------------------------------------------------------------
extern "C" void kernel_launch(void* const* d_in, const int* in_sizes, int n_in,
                              void* d_out, int out_size, void* d_ws, size_t ws_size,
                              hipStream_t stream) {
    const float* xt     = (const float*)d_in[0];   // [B,T,C]
    const float* W_ca   = (const float*)d_in[1];   // [C, 2C]
    const float* b_ca   = (const float*)d_in[2];   // [2C]
    const float* W_proj = (const float*)d_in[3];   // [C, C]
    const float* b_proj = (const float*)d_in[4];   // [C]
    float* out = (float*)d_out;

    char* ws = (char*)d_ws;
    size_t off = 0;
    uint16_t* xbf  = (uint16_t*)(ws + off); off += (size_t)MTOT * CDIM * 2;       // 8 MB
    uint16_t* qk   = (uint16_t*)(ws + off); off += (size_t)MTOT * CDIM * 2;       // 8 MB
    uint16_t* vt   = (uint16_t*)(ws + off); off += (size_t)MTOT * CDIM * 2;       // 8 MB
    uint16_t* ybf  = (uint16_t*)(ws + off); off += (size_t)MTOT * CDIM * 2;       // 8 MB
    uint16_t* wtca = (uint16_t*)(ws + off); off += (size_t)N2C  * CDIM * 2;       // 4 MB
    uint16_t* wtp  = (uint16_t*)(ws + off); off += (size_t)CDIM * CDIM * 2;       // 2 MB

    // prep
    {
        int n = MTOT * CDIM;
        k_convert_bf16<<<(n + 255) / 256, 256, 0, stream>>>(xt, xbf, n);
    }
    {
        int n = CDIM * N2C;
        k_transpose_bf16<<<(n + 255) / 256, 256, 0, stream>>>(W_ca, wtca, CDIM, N2C);
    }
    {
        int n = CDIM * CDIM;
        k_transpose_bf16<<<(n + 255) / 256, 256, 0, stream>>>(W_proj, wtp, CDIM, CDIM);
    }

    // qkv
    k_gemm_qkv<<<dim3(MTOT / 16, N2C / 256), 128, 0, stream>>>(xbf, wtca, b_ca, qk, vt);

    // attention
    k_attn<<<BSZ * NH * (TSEQ / 16), 32, 0, stream>>>(xbf, qk, vt, ybf);

    // projection
    k_gemm_proj<<<dim3(MTOT / 16, CDIM / 256), 128, 0, stream>>>(ybf, wtp, b_proj, out);
}
